// FlashCausalSelfAttention_90383291777499
// MI455X (gfx1250) — matrix operand order repaired and checked by reading, not compiled, not run on hardware
//
#include <hip/hip_runtime.h>
#include <hip/hip_bf16.h>
#include <math.h>

typedef __attribute__((ext_vector_type(16))) __bf16 v16bf;
typedef __attribute__((ext_vector_type(8)))  __bf16 v8bf;
typedef __attribute__((ext_vector_type(8)))  float  v8f;

#define B_DIM 8
#define T_DIM 1024
#define C_DIM 1024
#define H_DIM 16
#define D_DIM 64
#define BT    (B_DIM * T_DIM)   // 8192 rows
#define TC    (3 * C_DIM)       // 3072 qkv cols

__device__ __forceinline__ v8f wmma_bf16(v16bf a, v16bf b, v8f c) {
  // v_wmma_f32_16x16x32_bf16  (neg_a, A, neg_b, B, c_mod, C, reuse_a, reuse_b)
  return __builtin_amdgcn_wmma_f32_16x16x32_bf16(false, a, false, b, (short)0, c, false, false);
}

// Row of a 16x16 f32 C-tile lives on 16 contiguous lanes -> reduce within groups of 16.
__device__ __forceinline__ float rowmax16(float v) {
#pragma unroll
  for (int m = 8; m >= 1; m >>= 1) v = fmaxf(v, __shfl_xor(v, m, 16));
  return v;
}
__device__ __forceinline__ float rowsum16(float v) {
#pragma unroll
  for (int m = 8; m >= 1; m >>= 1) v += __shfl_xor(v, m, 16);
  return v;
}

// Load a 16-float run (fp32) and convert to one bf16 A/B fragment half-pair.
__device__ __forceinline__ v16bf cvt_frag_a(const float* g1, const float* g2) {
  // A fragment: elements 0..7 from g1[0..7], 8..15 from g2[0..7]
  float f[16];
  *(float4*)(f + 0)  = *(const float4*)(g1 + 0);
  *(float4*)(f + 4)  = *(const float4*)(g1 + 4);
  *(float4*)(f + 8)  = *(const float4*)(g2 + 0);
  *(float4*)(f + 12) = *(const float4*)(g2 + 4);
  v16bf a;
#pragma unroll
  for (int e = 0; e < 16; ++e) a[e] = (__bf16)f[e];
  return a;
}
__device__ __forceinline__ v16bf cvt_frag_b(const float* g) {
  // B fragment: 16 contiguous k values
  float f[16];
  *(float4*)(f + 0)  = *(const float4*)(g + 0);
  *(float4*)(f + 4)  = *(const float4*)(g + 4);
  *(float4*)(f + 8)  = *(const float4*)(g + 8);
  *(float4*)(f + 12) = *(const float4*)(g + 12);
  v16bf b;
#pragma unroll
  for (int e = 0; e < 16; ++e) b[e] = (__bf16)f[e];
  return b;
}

// ---------------------------------------------------------------------------
// Kernel 1: qkv = x @ W_attn^T  (fp32 in, bf16 out). 64x64 register tile per
// wave: 16 WMMAs per 8 fragment loads. Scatter q/k to (B,H,T,D), v to (B,H,D,T).
// x (32MB) + W_attn (12MB) stay resident in the 192MB L2.
// ---------------------------------------------------------------------------
__global__ __launch_bounds__(256)
void qkv_gemm_kernel(const float* __restrict__ x, const float* __restrict__ Wa,
                     __bf16* __restrict__ qbuf, __bf16* __restrict__ kbuf,
                     __bf16* __restrict__ vtbuf) {
  const int wid  = blockIdx.x * 8 + (threadIdx.x >> 5);   // 128*48 = 6144 waves
  const int lane = threadIdx.x & 31;
  const int half = lane >> 4;
  const int lidx = lane & 15;
  const int tn = wid % (TC / 64);
  const int tm = wid / (TC / 64);
  const int i0 = tm * 64, j0 = tn * 64;

  const float* arow[4];
  const float* brow[4];
#pragma unroll
  for (int mi = 0; mi < 4; ++mi)
    arow[mi] = x  + (size_t)(i0 + mi * 16 + lidx) * C_DIM + 16 * half;
#pragma unroll
  for (int nj = 0; nj < 4; ++nj)
    brow[nj] = Wa + (size_t)(j0 + nj * 16 + lidx) * C_DIM + 16 * half;

  v8f acc[4][4] = {};
  for (int kk = 0; kk < C_DIM; kk += 32) {
    v16bf a[4], b[4];
#pragma unroll
    for (int mi = 0; mi < 4; ++mi)
      a[mi] = cvt_frag_a(arow[mi] + kk, arow[mi] + kk + 8);
#pragma unroll
    for (int nj = 0; nj < 4; ++nj)
      b[nj] = cvt_frag_b(brow[nj] + kk);
#pragma unroll
    for (int mi = 0; mi < 4; ++mi)
#pragma unroll
      for (int nj = 0; nj < 4; ++nj)
        acc[mi][nj] = wmma_bf16(a[mi], b[nj], acc[mi][nj]);
  }

  // j0 is 64-aligned: one (sel, head) per 64-col tile.
  const int sel = j0 >> 10;                  // 0=q 1=k 2=v
  const int h   = (j0 & (C_DIM - 1)) >> 6;
#pragma unroll
  for (int mi = 0; mi < 4; ++mi) {
#pragma unroll
    for (int r = 0; r < 8; ++r) {
      const int i  = i0 + mi * 16 + r + 8 * half;
      const int b_ = i >> 10, t = i & (T_DIM - 1);
      const size_t bh = (size_t)(b_ * H_DIM + h);
#pragma unroll
      for (int nj = 0; nj < 4; ++nj) {
        const int d = nj * 16 + lidx;
        const __bf16 v = (__bf16)acc[mi][nj][r];
        if (sel == 0)      qbuf [(bh * T_DIM + t) * D_DIM + d] = v;
        else if (sel == 1) kbuf [(bh * T_DIM + t) * D_DIM + d] = v;
        else               vtbuf[(bh * D_DIM + d) * T_DIM + t] = v;
      }
    }
  }
}

// ---------------------------------------------------------------------------
// Kernel 2: faithful rotary with cos/sin of q itself, applied to q and k.
// One thread owns one (b,h,t) row of 64 -> in-place is race-free & deterministic.
// rot[d] = d<32 ? -x[2d+1] : x[2(d-32)]
// ---------------------------------------------------------------------------
__global__ __launch_bounds__(256)
void rotary_kernel(__bf16* __restrict__ qbuf, __bf16* __restrict__ kbuf) {
  const int rid = blockIdx.x * blockDim.x + threadIdx.x;   // [0, B*H*T)
  __bf16* qr = qbuf + (size_t)rid * D_DIM;
  __bf16* kr = kbuf + (size_t)rid * D_DIM;
  float qf[64], kf[64];
#pragma unroll
  for (int d = 0; d < 64; ++d) { qf[d] = (float)qr[d]; kf[d] = (float)kr[d]; }
#pragma unroll
  for (int d = 0; d < 64; ++d) {
    const int   pair = (d < 32) ? (2 * d + 1) : (2 * (d - 32));
    const float sgn  = (d < 32) ? -1.0f : 1.0f;
    const float c = __cosf(qf[d]);
    const float s = __sinf(qf[d]);
    qr[d] = (__bf16)(qf[d] * c + sgn * qf[pair] * s);
    kr[d] = (__bf16)(kf[d] * c + sgn * kf[pair] * s);
  }
}

// ---------------------------------------------------------------------------
// Kernel 3: flash attention, one wave per (b, h, 32-query tile).
// K/V fragments are loaded once per key block and reused by both 16-query
// halves: 16 WMMAs per 32-key block against 8 B-fragment loads.
// ---------------------------------------------------------------------------
__global__ __launch_bounds__(32)
void flash_attn_kernel(const __bf16* __restrict__ qbuf, const __bf16* __restrict__ kbuf,
                       const __bf16* __restrict__ vtbuf, __bf16* __restrict__ ybuf) {
  __shared__ __bf16 lp[2 * 16 * 32];   // P staging for both query halves
  const int bh   = blockIdx.x >> 5;
  const int qt   = blockIdx.x & 31;
  const int q0   = qt << 5;            // 32 queries per wave
  const int lane = threadIdx.x & 31;
  const int half = lane >> 4;
  const int lidx = lane & 15;
  const int b_ = bh >> 4, h = bh & 15;

  // Q A-fragments: 2 query halves x 2 K-steps over D=64
  v16bf aq[2][2];
#pragma unroll
  for (int qi = 0; qi < 2; ++qi) {
    const __bf16* qrow = qbuf + ((size_t)bh * T_DIM + q0 + qi * 16 + lidx) * D_DIM;
#pragma unroll
    for (int ds = 0; ds < 2; ++ds) {
      v8bf g0 = *(const v8bf*)(qrow + ds * 32 + 16 * half);
      v8bf g1 = *(const v8bf*)(qrow + ds * 32 + 16 * half + 8);
#pragma unroll
      for (int e = 0; e < 8; ++e) { aq[qi][ds][e] = g0[e]; aq[qi][ds][8 + e] = g1[e]; }
    }
  }

  v8f acc[2][4] = {};                  // 32 x 64 output accumulator
  float mrow[2][8], lrow[2][8];
#pragma unroll
  for (int qi = 0; qi < 2; ++qi)
#pragma unroll
    for (int r = 0; r < 8; ++r) { mrow[qi][r] = -1e30f; lrow[qi][r] = 0.0f; }

  const int nkb = qt + 1;              // key blocks of 32 covering keys 0..q0+31
  for (int jb = 0; jb < nkb; ++jb) {
    const int kb0 = jb << 5;

    // S (32x32): per query half, two 16-col tiles; K fragments shared.
    v8f s[2][2] = {};
#pragma unroll
    for (int ds = 0; ds < 2; ++ds) {
      const __bf16* kbase = kbuf + ((size_t)bh * T_DIM + kb0 + lidx) * D_DIM + ds * 32 + 16 * half;
      v16bf b0 = *(const v16bf*)(kbase);
      v16bf b1 = *(const v16bf*)(kbase + (size_t)16 * D_DIM);
#pragma unroll
      for (int qi = 0; qi < 2; ++qi) {
        s[qi][0] = wmma_bf16(aq[qi][ds], b0, s[qi][0]);
        s[qi][1] = wmma_bf16(aq[qi][ds], b1, s[qi][1]);
      }
    }

    // Online softmax update + write P tiles to LDS (bf16)
#pragma unroll
    for (int qi = 0; qi < 2; ++qi) {
#pragma unroll
      for (int r = 0; r < 8; ++r) {
        const int rowq = q0 + qi * 16 + r + 8 * half;
        float v0 = (kb0 + lidx      <= rowq) ? s[qi][0][r] : -1e30f;
        float v1 = (kb0 + 16 + lidx <= rowq) ? s[qi][1][r] : -1e30f;
        const float mc = rowmax16(fmaxf(v0, v1));
        const float mn = fmaxf(mrow[qi][r], mc);
        const float alpha = __expf(mrow[qi][r] - mn);
        const float p0 = __expf(v0 - mn);
        const float p1 = __expf(v1 - mn);
        lrow[qi][r] = lrow[qi][r] * alpha + rowsum16(p0 + p1);
        mrow[qi][r] = mn;
#pragma unroll
        for (int dt = 0; dt < 4; ++dt) acc[qi][dt][r] *= alpha;
        const int prow = r + 8 * half;
        lp[qi * 512 + prow * 32 + lidx]      = (__bf16)p0;
        lp[qi * 512 + prow * 32 + 16 + lidx] = (__bf16)p1;
      }
    }
    asm volatile("s_wait_dscnt 0x0" ::: "memory");   // LDS writes visible before A-frag reads

    // P as A-fragments (row m = lidx, k = 16h+0..15)
    v16bf pa[2];
#pragma unroll
    for (int qi = 0; qi < 2; ++qi)
#pragma unroll
      for (int e = 0; e < 8; ++e) {
        pa[qi][e]     = lp[qi * 512 + lidx * 32 + 16 * half + e];
        pa[qi][8 + e] = lp[qi * 512 + lidx * 32 + 16 * half + 8 + e];
      }

    // acc += P (32x32) @ V (32x64): V^T fragments shared by both query halves.
#pragma unroll
    for (int dt = 0; dt < 4; ++dt) {
      const __bf16* vbase = vtbuf + ((size_t)bh * D_DIM + dt * 16 + lidx) * T_DIM + kb0 + 16 * half;
      v16bf bv = *(const v16bf*)(vbase);
#pragma unroll
      for (int qi = 0; qi < 2; ++qi)
        acc[qi][dt] = wmma_bf16(pa[qi], bv, acc[qi][dt]);
    }
    asm volatile("s_wait_dscnt 0x0" ::: "memory");   // reads done before next block's writes
  }

  // y[(b*T + q)*C + h*64 + d] = acc / l  (bf16 scratch feeding the proj GEMM)
#pragma unroll
  for (int qi = 0; qi < 2; ++qi)
#pragma unroll
    for (int r = 0; r < 8; ++r) {
      const float inv = 1.0f / lrow[qi][r];
      const size_t row = (size_t)(b_ * T_DIM + q0 + qi * 16 + r + 8 * half) * C_DIM + h * D_DIM;
#pragma unroll
      for (int dt = 0; dt < 4; ++dt)
        ybuf[row + dt * 16 + lidx] = (__bf16)(acc[qi][dt][r] * inv);
    }
}

// ---------------------------------------------------------------------------
// Kernel 4: out = y @ W_proj^T  (bf16 x fp32->bf16 -> fp32 out), 64x64 tile.
// ---------------------------------------------------------------------------
__global__ __launch_bounds__(256)
void proj_gemm_kernel(const __bf16* __restrict__ ybuf, const float* __restrict__ Wp,
                      float* __restrict__ out) {
  const int wid  = blockIdx.x * 8 + (threadIdx.x >> 5);   // 128*16 = 2048 waves
  const int lane = threadIdx.x & 31;
  const int half = lane >> 4;
  const int lidx = lane & 15;
  const int tn = wid & 15;
  const int tm = wid >> 4;
  const int i0 = tm * 64, j0 = tn * 64;

  const __bf16* arow[4];
  const float*  brow[4];
#pragma unroll
  for (int mi = 0; mi < 4; ++mi)
    arow[mi] = ybuf + (size_t)(i0 + mi * 16 + lidx) * C_DIM + 16 * half;
#pragma unroll
  for (int nj = 0; nj < 4; ++nj)
    brow[nj] = Wp + (size_t)(j0 + nj * 16 + lidx) * C_DIM + 16 * half;

  v8f acc[4][4] = {};
  for (int kk = 0; kk < C_DIM; kk += 32) {
    v16bf a[4], b[4];
#pragma unroll
    for (int mi = 0; mi < 4; ++mi) {
      v8bf g0 = *(const v8bf*)(arow[mi] + kk);
      v8bf g1 = *(const v8bf*)(arow[mi] + kk + 8);
#pragma unroll
      for (int e = 0; e < 8; ++e) { a[mi][e] = g0[e]; a[mi][8 + e] = g1[e]; }
    }
#pragma unroll
    for (int nj = 0; nj < 4; ++nj)
      b[nj] = cvt_frag_b(brow[nj] + kk);
#pragma unroll
    for (int mi = 0; mi < 4; ++mi)
#pragma unroll
      for (int nj = 0; nj < 4; ++nj)
        acc[mi][nj] = wmma_bf16(a[mi], b[nj], acc[mi][nj]);
  }

#pragma unroll
  for (int mi = 0; mi < 4; ++mi)
#pragma unroll
    for (int r = 0; r < 8; ++r) {
      const size_t row = (size_t)(i0 + mi * 16 + r + 8 * half) * C_DIM;
#pragma unroll
      for (int nj = 0; nj < 4; ++nj)
        out[row + j0 + nj * 16 + lidx] = acc[mi][nj][r];
    }
}

// ---------------------------------------------------------------------------
extern "C" void kernel_launch(void* const* d_in, const int* in_sizes, int n_in,
                              void* d_out, int out_size, void* d_ws, size_t ws_size,
                              hipStream_t stream) {
  (void)in_sizes; (void)n_in; (void)out_size; (void)ws_size;
  const float* x  = (const float*)d_in[0];   // (8,1024,1024)
  const float* Wa = (const float*)d_in[1];   // (3072,1024)
  const float* Wp = (const float*)d_in[2];   // (1024,1024)
  float* out = (float*)d_out;                // (8,1024,1024) fp32

  // Workspace layout (bf16): q, k (B,H,T,D); v^T (B,H,D,T); y (B*T, C). 64 MB total.
  const size_t QSZ = (size_t)B_DIM * H_DIM * T_DIM * D_DIM;   // 8 Mi elements
  __bf16* qbuf  = (__bf16*)d_ws;
  __bf16* kbuf  = qbuf  + QSZ;
  __bf16* vtbuf = kbuf  + QSZ;
  __bf16* ybuf  = vtbuf + QSZ;

  // 128 x 48 64x64-tiles, 8 waves/block
  qkv_gemm_kernel<<<768, 256, 0, stream>>>(x, Wa, qbuf, kbuf, vtbuf);
  // B*H*T = 131072 rows
  rotary_kernel<<<512, 256, 0, stream>>>(qbuf, kbuf);
  // B*H*(T/32) = 4096 waves
  flash_attn_kernel<<<4096, 32, 0, stream>>>(qbuf, kbuf, vtbuf, ybuf);
  // 128 x 16 64x64-tiles, 8 waves/block
  proj_gemm_kernel<<<256, 256, 0, stream>>>(ybuf, Wp, out);
}
